// WorldModel_16870631539022
// MI455X (gfx1250) — matrix operand order repaired and checked by reading, not compiled, not run on hardware
//
#include <hip/hip_runtime.h>
#include <hip/hip_bf16.h>
#include <math.h>

// ---------------------------------------------------------------------------
// RSSM world model for MI455X (gfx1250, wave32, WMMA bf16):
//  Phase 0: weight convert/transpose to bf16 (tiny).
//  Phase 1: batch-parallel GEMM over all B*T rows (register-blocked, 7 C tiles
//           per wave; kt-outer so each A frag is ds_load'ed exactly once;
//           B frags batch-loaded so WMMAs drain a pipeline of in-flight loads).
//  Phase 2: persistent recurrent kernel, 64 WGs x 16 batch rows, t-loop inside;
//           gi/qe rows async-copied global->LDS (ASYNCcnt) overlapping the
//           gh GEMM on the serial critical path.
// ---------------------------------------------------------------------------

typedef __attribute__((ext_vector_type(16))) __bf16 v16bf;
typedef __attribute__((ext_vector_type(8)))  __bf16 v8bf;
typedef __attribute__((ext_vector_type(8)))  float  v8f;
typedef __attribute__((ext_vector_type(4)))  int    v4i;

#define B_   1024
#define T_   50
#define E_   1024
#define A_   6
#define S_   200
#define L_   30
#define NCAT 816    // 608 (gi, padded from 600) + 208 (qe, padded from 200)
#define KPS  224    // K=200 padded to multiple of 32
#define OUTC 350    // S + 5L

#if defined(__has_builtin)
#if __has_builtin(__builtin_amdgcn_global_load_async_to_lds_b128) && \
    __has_builtin(__builtin_amdgcn_s_wait_asynccnt)
#define HAVE_ASYNC_LDS 1
#endif
#endif
#ifndef HAVE_ASYNC_LDS
#define HAVE_ASYNC_LDS 0
#endif

__device__ __forceinline__ float softplusf_(float x) {
  return (x > 20.f) ? x : log1pf(expf(x));
}
__device__ __forceinline__ float eluf_(float x) {
  return (x > 0.f) ? x : expm1f(x);
}

// A-frag (16x32 bf16, row-major source): lanes 0-15 = rows; half selects K group.
__device__ __forceinline__ v16bf load_a_frag(const __bf16* base, int ldk, int m, int lh, int kb) {
  const __bf16* p = base + (size_t)m * ldk + kb + lh * 16;
  v8bf c0 = *(const v8bf*)(p);
  v8bf c1 = *(const v8bf*)(p + 8);
  v16bf a;
#pragma unroll
  for (int i = 0; i < 8; ++i) { a[i] = c0[i]; a[i + 8] = c1[i]; }
  return a;
}

// B-frag (32x16 bf16) from pre-transposed W^T [N][Kpad] row-major.
__device__ __forceinline__ v16bf load_b_frag(const __bf16* wt, int ldk, int ncol, int lh, int kb) {
  return *(const v16bf*)(wt + (size_t)ncol * ldk + kb + lh * 16);
}

// 16-byte global -> LDS copy: async on CDNA5 (tracked by ASYNCcnt), sync fallback.
__device__ __forceinline__ void cp16_g2l(const float* g, float* l) {
#if HAVE_ASYNC_LDS
  __builtin_amdgcn_global_load_async_to_lds_b128(
      (__attribute__((address_space(1))) v4i*)g,
      (__attribute__((address_space(3))) v4i*)l, 0, 0);
#else
  *(float4*)l = *(const float4*)g;
#endif
}

// ---------------- Phase 0: weight convert + transpose + zero-pad -----------
__global__ __launch_bounds__(256) void k_cvt_t(const float* __restrict__ src,
                                               __bf16* __restrict__ dst,
                                               int dstN, int dstK, int nUse, int kUse,
                                               int srcLD, int rowOff) {
  int idx = blockIdx.x * 256 + threadIdx.x;
  if (idx >= dstN * dstK) return;
  int n = idx / dstK, k = idx - n * dstK;
  float v = 0.f;
  if (n < nUse && k < kUse) v = src[(size_t)(k + rowOff) * srcLD + n];
  dst[idx] = (__bf16)v;
}

// ---------------- Phase 1: GIQE = [gi | qe] over all 51200 rows ------------
__global__ __launch_bounds__(256) void k_giqe(const float* __restrict__ embed,
                                              const float* __restrict__ action,
                                              const float* __restrict__ W_ih,
                                              const float* __restrict__ b_ih,
                                              const float* __restrict__ bq1,
                                              const __bf16* __restrict__ WcatT,
                                              float* __restrict__ GIQE) {
  __shared__ __align__(16) __bf16 sA[16 * 1024];
  __shared__ float sAct[16 * 8];
  const int bt0 = blockIdx.x * 16;
  const int tid = threadIdx.x;

  // Stage 16 rows of embed (fp32 -> bf16) into LDS once, reused by all N-tiles.
  for (int i = tid; i < 16 * 1024; i += 256) {
    int m = i >> 10, k = i & 1023;
    sA[i] = (__bf16)embed[(size_t)(bt0 + m) * E_ + k];
  }
  for (int i = tid; i < 16 * 6; i += 256) {
    int m = i / 6, k = i - m * 6;
    sAct[m * 8 + k] = action[(size_t)(bt0 + m) * A_ + k];
  }
  __syncthreads();

  const int wv = tid >> 5, lane = tid & 31, lh = lane >> 4, mh = lane & 15;

  // Register blocking: 7 C tiles per wave; kt-outer so each A frag loads once.
  int ncols[7];
#pragma unroll
  for (int i = 0; i < 7; ++i) {
    int nt = wv + 8 * i; if (nt > 50) nt = 50;   // clamp: duplicate work, no store
    ncols[i] = nt * 16 + mh;
  }
  v8f acc[7];
#pragma unroll
  for (int i = 0; i < 7; ++i)
#pragma unroll
    for (int rr = 0; rr < 8; ++rr) acc[i][rr] = 0.f;

  for (int kt = 0; kt < 32; ++kt) {
    v16bf a = load_a_frag(sA, 1024, mh, lh, kt * 32);
    // Batch-issue all 7 B-frag loads (one clause), then drain through WMMAs so
    // later loads stay in flight while earlier tiles compute.
    v16bf bfr[7];
#pragma unroll
    for (int i = 0; i < 7; ++i) bfr[i] = load_b_frag(WcatT, 1024, ncols[i], lh, kt * 32);
#pragma unroll
    for (int i = 0; i < 7; ++i)
      acc[i] = __builtin_amdgcn_wmma_f32_16x16x32_bf16(false, a, false, bfr[i], (short)0, acc[i], false, false);
  }

#pragma unroll
  for (int i = 0; i < 7; ++i) {
    if (wv + 8 * i > 50) continue;
    const int ncol = ncols[i];
    float bias = 0.f;
    if (ncol < 600) bias = b_ih[ncol];
    else if (ncol >= 608 && ncol < 808) bias = bq1[ncol - 608];
    float wact[6];
    const bool hasAct = (ncol < 600);
    if (hasAct) {
#pragma unroll
      for (int k = 0; k < 6; ++k) wact[k] = W_ih[(size_t)k * 600 + ncol];
    }
#pragma unroll
    for (int rr = 0; rr < 8; ++rr) {
      const int m2 = rr + lh * 8;
      float v = acc[i][rr] + bias;
      if (hasAct) {
#pragma unroll
        for (int k = 0; k < 6; ++k) v += sAct[m2 * 8 + k] * wact[k];  // K=6 action tail
      }
      GIQE[(size_t)(bt0 + m2) * NCAT + ncol] = v;
    }
  }
}

// ---------------- Phase 2: persistent recurrent kernel ---------------------
__global__ __launch_bounds__(256) void k_rssm(const float* __restrict__ noise,
                                              const float* __restrict__ b_hh,
                                              const float* __restrict__ bp1,
                                              const float* __restrict__ bp2,
                                              const float* __restrict__ bq2,
                                              const __bf16* __restrict__ WhhT,
                                              const __bf16* __restrict__ Wp1T,
                                              const __bf16* __restrict__ Wq1dT,
                                              const __bf16* __restrict__ Wp2T,
                                              const __bf16* __restrict__ Wq2T,
                                              const float* __restrict__ GIQE,
                                              float* __restrict__ out) {
  __shared__ __align__(16) __bf16 sH[16 * KPS];    // h state, bf16 (WMMA A operand)
  __shared__ __align__(16) __bf16 sPH[16 * KPS];   // prior hidden (post-ELU)
  __shared__ __align__(16) __bf16 sQH[16 * KPS];   // posterior hidden (post-ELU)
  __shared__ __align__(16) float sHf[16 * 208];    // h state, fp32 (exact recurrence)
  __shared__ __align__(16) float sGH[16 * 608];    // h @ W_hh + b_hh
  __shared__ __align__(16) float sGI[16 * 600];    // gi rows for this t (async staged)
  __shared__ __align__(16) float sQE[16 * 200];    // qe rows for this t (async staged)
  __shared__ __align__(16) float sP2[16 * 64];     // prior head out
  __shared__ __align__(16) float sQ2[16 * 64];     // posterior head out

  const int b0 = blockIdx.x * 16;
  const int tid = threadIdx.x;
  const int wv = tid >> 5, lane = tid & 31, lh = lane >> 4, mh = lane & 15;

  for (int i = tid; i < 16 * KPS; i += 256) {
    sH[i] = (__bf16)0.f; sPH[i] = (__bf16)0.f; sQH[i] = (__bf16)0.f;
  }
  for (int i = tid; i < 16 * 208; i += 256) sHf[i] = 0.f;
  __syncthreads();

  for (int t = 0; t < T_; ++t) {
    // -- Issue async global->LDS copies of gi (16x600) and qe (16x200) rows.
    //    These overlap the gh GEMM below; completion enforced before Stage 2.
    for (int c = tid; c < 3200; c += 256) {   // 3200 x 16B chunks
      if (c < 2400) {
        const int m = c / 150, cc = c - m * 150;
        cp16_g2l(GIQE + ((size_t)(b0 + m) * T_ + t) * NCAT + cc * 4,
                 sGI + m * 600 + cc * 4);
      } else {
        const int c2 = c - 2400;
        const int m = c2 / 50, cc = c2 - m * 50;
        cp16_g2l(GIQE + ((size_t)(b0 + m) * T_ + t) * NCAT + 608 + cc * 4,
                 sQE + m * 200 + cc * 4);
      }
    }

    // -- Stage 1: gh = h @ W_hh + b_hh (separate from gi: r-gate coupling).
    //    Preload all 7 h fragments once; batch-load B frags per tile so the
    //    serial accumulate chain overlaps the remaining in-flight loads.
    {
      v16bf ah[7];
#pragma unroll
      for (int kt = 0; kt < 7; ++kt) ah[kt] = load_a_frag(sH, KPS, mh, lh, kt * 32);
      for (int nt = wv; nt < 38; nt += 8) {
        const int ncol = nt * 16 + mh;
        const float bias = (ncol < 600) ? b_hh[ncol] : 0.f;
        v16bf bfr[7];
#pragma unroll
        for (int kt = 0; kt < 7; ++kt) bfr[kt] = load_b_frag(WhhT, KPS, ncol, lh, kt * 32);
        v8f c;
#pragma unroll
        for (int rr = 0; rr < 8; ++rr) c[rr] = bias;
#pragma unroll
        for (int kt = 0; kt < 7; ++kt)
          c = __builtin_amdgcn_wmma_f32_16x16x32_bf16(false, ah[kt], false, bfr[kt], (short)0, c, false, false);
#pragma unroll
        for (int rr = 0; rr < 8; ++rr) sGH[(rr + lh * 8) * 608 + ncol] = c[rr];
      }
    }
#if HAVE_ASYNC_LDS
    __builtin_amdgcn_s_wait_asynccnt(0);
#endif
    __syncthreads();

    // -- Stage 2: GRU gate fusion (fp32), h update, write deter output.
    for (int i = tid; i < 16 * 200; i += 256) {
      const int m = i / 200, j = i - m * 200;
      const float rg = 1.f / (1.f + expf(-(sGI[m * 600 + j]       + sGH[m * 608 + j])));
      const float zg = 1.f / (1.f + expf(-(sGI[m * 600 + j + 200] + sGH[m * 608 + j + 200])));
      const float ng = tanhf(sGI[m * 600 + j + 400] + rg * sGH[m * 608 + j + 400]);
      const float hprev = sHf[m * 208 + j];
      const float hnew = (1.f - zg) * ng + zg * hprev;
      sHf[m * 208 + j] = hnew;
      sH[m * KPS + j] = (__bf16)hnew;
      out[((size_t)(b0 + m) * T_ + t) * OUTC + j] = hnew;       // deter
    }
    __syncthreads();

    // -- Stage 3: prior hidden = ELU(h@Wp1+bp1); posterior = ELU(h@Wq1d + qe).
    {
      v16bf ah[7];
#pragma unroll
      for (int kt = 0; kt < 7; ++kt) ah[kt] = load_a_frag(sH, KPS, mh, lh, kt * 32);
      for (int id = wv; id < 26; id += 8) {
        const bool pri = id < 13;
        const int nt = pri ? id : id - 13;
        const int ncol = nt * 16 + mh;                   // < 208
        const __bf16* Wt = pri ? Wp1T : Wq1dT;
        v16bf bfr[7];
#pragma unroll
        for (int kt = 0; kt < 7; ++kt) bfr[kt] = load_b_frag(Wt, KPS, ncol, lh, kt * 32);
        v8f c;
        if (pri) {
          const float bias = (ncol < 200) ? bp1[ncol] : 0.f;
#pragma unroll
          for (int rr = 0; rr < 8; ++rr) c[rr] = bias;
        } else {
#pragma unroll
          for (int rr = 0; rr < 8; ++rr) {
            const int m2 = rr + lh * 8;
            c[rr] = (ncol < 200) ? sQE[m2 * 200 + ncol] : 0.f;  // e@Wq1[S:]+bq1
          }
        }
#pragma unroll
        for (int kt = 0; kt < 7; ++kt)
          c = __builtin_amdgcn_wmma_f32_16x16x32_bf16(false, ah[kt], false, bfr[kt], (short)0, c, false, false);
        __bf16* dst = pri ? sPH : sQH;
#pragma unroll
        for (int rr = 0; rr < 8; ++rr) dst[(rr + lh * 8) * KPS + ncol] = (__bf16)eluf_(c[rr]);
      }
    }
    __syncthreads();

    // -- Stage 4: head GEMMs (exactly one 16x16 tile per wave).
    {
      const bool pri = wv < 4;
      const int nt = pri ? wv : wv - 4;
      const int ncol = nt * 16 + mh;                     // < 64
      const __bf16* Wt = pri ? Wp2T : Wq2T;
      const __bf16* As = pri ? sPH : sQH;
      const float* bs = pri ? bp2 : bq2;
      v16bf bfr[7];
#pragma unroll
      for (int kt = 0; kt < 7; ++kt) bfr[kt] = load_b_frag(Wt, KPS, ncol, lh, kt * 32);
      v8f c;
      const float bias = (ncol < 60) ? bs[ncol] : 0.f;
#pragma unroll
      for (int rr = 0; rr < 8; ++rr) c[rr] = bias;
#pragma unroll
      for (int kt = 0; kt < 7; ++kt) {
        v16bf a = load_a_frag(As, KPS, mh, lh, kt * 32);
        c = __builtin_amdgcn_wmma_f32_16x16x32_bf16(false, a, false, bfr[kt], (short)0, c, false, false);
      }
      float* dst = pri ? sP2 : sQ2;
#pragma unroll
      for (int rr = 0; rr < 8; ++rr) dst[(rr + lh * 8) * 64 + ncol] = c[rr];
    }
    __syncthreads();

    // -- Stage 5: softplus, rsample, scatter outputs.
    for (int i = tid; i < 16 * L_; i += 256) {
      const int m = i / L_, l = i - m * L_;
      const size_t bt = (size_t)(b0 + m) * T_ + t;
      const float pm = sP2[m * 64 + l];
      const float ps = softplusf_(sP2[m * 64 + l + 30]) + 0.1f;
      const float qm = sQ2[m * 64 + l];
      const float qs = softplusf_(sQ2[m * 64 + l + 30]) + 0.1f;
      const float eps = noise[bt * L_ + l];
      float* o = out + bt * OUTC;
      o[200 + l] = qm + qs * eps;   // stoch
      o[230 + l] = pm;
      o[260 + l] = ps;
      o[290 + l] = qm;
      o[320 + l] = qs;
    }
    __syncthreads();
  }
}

// ---------------------------------------------------------------------------
extern "C" void kernel_launch(void* const* d_in, const int* in_sizes, int n_in,
                              void* d_out, int out_size, void* d_ws, size_t ws_size,
                              hipStream_t stream) {
  (void)in_sizes; (void)n_in; (void)out_size; (void)ws_size;
  const float* action = (const float*)d_in[0];
  const float* embed  = (const float*)d_in[1];
  const float* noise  = (const float*)d_in[2];
  const float* W_ih   = (const float*)d_in[3];
  const float* W_hh   = (const float*)d_in[4];
  const float* b_ih   = (const float*)d_in[5];
  const float* b_hh   = (const float*)d_in[6];
  const float* Wp1    = (const float*)d_in[7];
  const float* bp1    = (const float*)d_in[8];
  const float* Wp2    = (const float*)d_in[9];
  const float* bp2    = (const float*)d_in[10];
  const float* Wq1    = (const float*)d_in[11];
  const float* bq1    = (const float*)d_in[12];
  const float* Wq2    = (const float*)d_in[13];
  const float* bq2    = (const float*)d_in[14];

  char* ws = (char*)d_ws;
  size_t off = 0;
  auto take = [&](size_t bytes) {
    char* p = ws + off;
    off = (off + bytes + 255) & ~(size_t)255;
    return p;
  };
  __bf16* WcatT = (__bf16*)take((size_t)816 * 1024 * 2);  // [gi 608 | qe 208] x K=1024
  __bf16* WhhT  = (__bf16*)take((size_t)608 * 224 * 2);
  __bf16* Wq1dT = (__bf16*)take((size_t)208 * 224 * 2);
  __bf16* Wp1T  = (__bf16*)take((size_t)208 * 224 * 2);
  __bf16* Wp2T  = (__bf16*)take((size_t)64 * 224 * 2);
  __bf16* Wq2T  = (__bf16*)take((size_t)64 * 224 * 2);
  float*  GIQE  = (float*)take((size_t)51200 * NCAT * 4); // ~167 MB

  auto cvt = [&](const float* src, __bf16* dst, int dN, int dK, int nU, int kU, int ld, int ro) {
    int tot = dN * dK;
    k_cvt_t<<<(tot + 255) / 256, 256, 0, stream>>>(src, dst, dN, dK, nU, kU, ld, ro);
  };
  // W^T layouts, zero-padded. WcatT rows 0..607 = W_ih[6:,:]^T, rows 608..815 = Wq1[200:,:]^T.
  cvt(W_ih, WcatT,            608, 1024, 600, 1024, 600, 6);
  cvt(Wq1,  WcatT + 608*1024, 208, 1024, 200, 1024, 200, 200);
  cvt(W_hh, WhhT,  608, 224, 600, 200, 600, 0);
  cvt(Wq1,  Wq1dT, 208, 224, 200, 200, 200, 0);
  cvt(Wp1,  Wp1T,  208, 224, 200, 200, 200, 0);
  cvt(Wp2,  Wp2T,   64, 224,  60, 200,  60, 0);
  cvt(Wq2,  Wq2T,   64, 224,  60, 200,  60, 0);

  // Phase 1: 3200 row-tiles of 16; each wave owns 7 register-blocked C tiles.
  k_giqe<<<3200, 256, 0, stream>>>(embed, action, W_ih, b_ih, bq1, WcatT, GIQE);

  // Phase 2: 64 persistent WGs, 16 batch rows each, t-loop inside.
  k_rssm<<<64, 256, 0, stream>>>(noise, b_hh, bp1, bp2, bq2,
                                 WhhT, Wp1T, Wq1dT, Wp2T, Wq2T, GIQE, (float*)d_out);
}
